// HyperbolicMHA_41317585387573
// MI455X (gfx1250) — matrix operand order repaired and checked by reading, not compiled, not run on hardware
//
#include <hip/hip_runtime.h>

typedef _Float16 half_t;
typedef __attribute__((ext_vector_type(16))) _Float16 v16h;
typedef __attribute__((ext_vector_type(8)))  _Float16 v8h;
typedef __attribute__((ext_vector_type(8)))  float    v8f;
typedef __attribute__((ext_vector_type(4)))  unsigned int u32x4;
typedef __attribute__((ext_vector_type(8)))  int          i32x8;
typedef __attribute__((ext_vector_type(4)))  int          i32x4;

// Problem constants (B=2, S=2048, D_MODEL=512, H=8, DK=DV=32)
#define BB   2
#define SS   2048
#define DM   512
#define HH   8
#define HD   256            // H*DK
#define NROW (BB*SS)        // 4096 rows of the projection matrices
#define HEADROWS (BB*HH*SS) // 32768 per-head rows of 32

#define HAVE_TDM __has_builtin(__builtin_amdgcn_tensor_load_to_lds)

static __device__ __forceinline__ v16h cat8(v8h lo, v8h hi) {
  return __builtin_shufflevector(lo, hi, 0,1,2,3,4,5,6,7,8,9,10,11,12,13,14,15);
}

#if HAVE_TDM
// ---------------------------------------------------------------------------
// Tensor Data Mover: 2-D tile load (16-bit elements) global -> LDS.
// D# built per CDNA5 ISA 8.3/8.4: group0 {count=1, lds_addr, global_addr,
// type=2}; group1 {data_size=1(2B), tensor_dim0/1, tile_dim0/1, dim0 stride}.
// Issued by one wave, tracked with TENSORcnt.
// ---------------------------------------------------------------------------
static __device__ __forceinline__ void tdm_load_2d_f16(
    unsigned lds_off, const void* gptr,
    unsigned tensor_d0, unsigned tensor_d1,
    unsigned tile_d0, unsigned tile_d1, unsigned stride0_elems) {
  unsigned long long ga = (unsigned long long)(uintptr_t)gptr;
  u32x4 g0;
  g0[0] = 1u;                                   // count=1 (valid), no gather
  g0[1] = lds_off;                              // LDS byte address
  g0[2] = (unsigned)(ga & 0xffffffffu);         // global_addr[31:0]
  g0[3] = (unsigned)(ga >> 32) | (2u << 30);    // global_addr[56:32] | type=2
  i32x8 g1;
  g1[0] = (int)(1u << 16);                      // wg_mask=0, data_size=1 (2B)
  g1[1] = (int)((tensor_d0 & 0xffffu) << 16);   // tensor_dim0[15:0] @ [63:48]
  g1[2] = (int)((tensor_d0 >> 16) | ((tensor_d1 & 0xffffu) << 16));
  g1[3] = (int)((tensor_d1 >> 16) | (tile_d0 << 16));   // tile_dim0 @ [127:112]
  g1[4] = (int)(tile_d1 & 0xffffu);             // tile_dim1, tile_dim2=0
  g1[5] = (int)stride0_elems;                   // tensor_dim0_stride[31:0]
  g1[6] = 0;                                    // stride0[47:32]=0, stride1 lo=0
  g1[7] = 0;                                    // stride1 hi = 0
  i32x4 z4 = {0, 0, 0, 0};
#if __clang_major__ >= 23
  i32x8 z8 = {0, 0, 0, 0, 0, 0, 0, 0};
  __builtin_amdgcn_tensor_load_to_lds(g0, g1, z4, z4, z8, 0);
#else
  __builtin_amdgcn_tensor_load_to_lds(g0, g1, z4, z4, 0);
#endif
}
#endif

// ---------------------------------------------------------------------------
// Weight transpose + f16 convert:  WT[c*R + r] = (f16) W[r*C + c]
// ---------------------------------------------------------------------------
__global__ void wtrans_kernel(const float* __restrict__ W, half_t* __restrict__ WT,
                              int R, int C) {
  int i = blockIdx.x * blockDim.x + threadIdx.x;
  if (i >= R * C) return;
  int r = i / C, c = i % C;
  WT[(size_t)c * R + r] = (half_t)W[i];
}

// ---------------------------------------------------------------------------
// QKV projection: P = X @ W^T + bias   ([4096,512] x [512,256] -> f16 [4096,256])
// grid = (64, 4, 3) blocks of 128 threads; wave computes a 16x64 strip.
// ---------------------------------------------------------------------------
__global__ void __launch_bounds__(128)
gemm_proj_kernel(const float* __restrict__ Xq, const float* __restrict__ Xk,
                 const float* __restrict__ Xv,
                 const half_t* __restrict__ WTq, const half_t* __restrict__ WTk,
                 const half_t* __restrict__ WTv,
                 const float* __restrict__ bq, const float* __restrict__ bk,
                 const float* __restrict__ bv,
                 half_t* __restrict__ Pq, half_t* __restrict__ Pk,
                 half_t* __restrict__ Pv) {
  const float*  X;  const half_t* WT;  const float* bias;  half_t* P;
  int z = blockIdx.z;
  if (z == 0)      { X = Xq; WT = WTq; bias = bq; P = Pq; }
  else if (z == 1) { X = Xk; WT = WTk; bias = bk; P = Pk; }
  else             { X = Xv; WT = WTv; bias = bv; P = Pv; }

  int tid = threadIdx.x, wave = tid >> 5, lane = tid & 31;
  int mm = lane & 15, hg = lane >> 4;
  int m0 = blockIdx.x * 64 + wave * 16;
  int n0 = blockIdx.y * 64;

  v8f acc[4] = {};
  for (int k0 = 0; k0 < DM; k0 += 32) {
    const float* ar = X + (size_t)(m0 + mm) * DM + k0;
    v16h a;
#pragma unroll
    for (int i = 0; i < 8; ++i) {
      a[i]     = (half_t)ar[8 * hg + i];
      a[i + 8] = (half_t)ar[16 + 8 * hg + i];
    }
    const half_t* br = WT + (size_t)k0 * HD + n0 + mm;
#pragma unroll
    for (int t = 0; t < 4; ++t) {
      v16h b;
#pragma unroll
      for (int i = 0; i < 8; ++i) {
        b[i]     = br[(size_t)(8 * hg + i) * HD + t * 16];
        b[i + 8] = br[(size_t)(16 + 8 * hg + i) * HD + t * 16];
      }
      acc[t] = __builtin_amdgcn_wmma_f32_16x16x32_f16(false, a, false, b,
                                                      (short)0, acc[t], false, false);
    }
  }
#pragma unroll
  for (int t = 0; t < 4; ++t) {
    int col = n0 + t * 16 + mm;
    float bb = bias[col];
#pragma unroll
    for (int r = 0; r < 8; ++r) {
      int row = m0 + r + 8 * hg;                 // C layout: M = r + 8*hg
      P[(size_t)row * HD + col] = (half_t)(acc[t][r] + bb);
    }
  }
}

// ---------------------------------------------------------------------------
// Per-head-row norms: t = sqrt(1 + |row32|^2) for q0 / k0 / t_v
// ---------------------------------------------------------------------------
__global__ void aux_norm_kernel(const half_t* __restrict__ Pq,
                                const half_t* __restrict__ Pk,
                                const half_t* __restrict__ Pv,
                                float* __restrict__ q0, float* __restrict__ k0,
                                float* __restrict__ tv) {
  int g = blockIdx.x * blockDim.x + threadIdx.x;   // [0, 3*32768)
  int sel = g >> 15, row = g & (HEADROWS - 1);
  const half_t* P = (sel == 0) ? Pq : (sel == 1) ? Pk : Pv;
  const half_t* pr = P + (size_t)row * 32;
  float s = 1.0f;
#pragma unroll
  for (int i = 0; i < 32; ++i) { float v = (float)pr[i]; s = fmaf(v, v, s); }
  float t = __builtin_sqrtf(s);
  ((sel == 0) ? q0 : (sel == 1) ? k0 : tv)[row] = t;
}

// ---------------------------------------------------------------------------
// Per-head transpose of K projection: Kt[bh][d][s] = Pk[bh][s][d]
// ---------------------------------------------------------------------------
__global__ void ktrans_kernel(const half_t* __restrict__ Pk, half_t* __restrict__ Kt) {
  int i = blockIdx.x * blockDim.x + threadIdx.x;   // [0, 16*65536)
  int s = i & 2047, d = (i >> 11) & 31, bh = i >> 16;
  Kt[(size_t)bh * 65536 + (size_t)d * 2048 + s] =
      Pk[(size_t)bh * 65536 + (size_t)s * 32 + d];
}

// ---------------------------------------------------------------------------
// Fused hyperbolic attention.
// grid = (S/64, B*H), block = 128 (4 waves); each wave owns 16 q rows.
// K/V tiles (64 keys) are DMA'd to LDS by the Tensor Data Mover (wave 0),
// double-buffered, drained with s_wait_tensorcnt; all 4 waves share them.
// Per tile: 4 score WMMAs -> arccosh/exp -> LDS relayout -> 4 midpoint WMMAs.
// ---------------------------------------------------------------------------
__global__ void __launch_bounds__(128)
attn_kernel(const half_t* __restrict__ Pq, const half_t* __restrict__ Kt,
            const half_t* __restrict__ Pv,
            const float* __restrict__ q0a, const float* __restrict__ k0a,
            const float* __restrict__ tva,
            const float* __restrict__ betap, const float* __restrict__ cparp,
            half_t* __restrict__ res) {
  __shared__ half_t ktile[2][32 * 64];   // [d][key] per buffer (8 KB)
  __shared__ half_t vtile[2][64 * 32];   // [key][vd] per buffer (8 KB)
  __shared__ half_t atile[4][16 * 64];   // per-wave alpha staging (8 KB)

  int tid = threadIdx.x, wave = tid >> 5, lane = tid & 31;
  int mm = lane & 15, hg = lane >> 4;
  int bh = blockIdx.y, b = bh >> 3, h = bh & 7;

  const half_t* Qh  = Pq + (size_t)bh * 65536;   // [2048][32] natural
  const half_t* Kth = Kt + (size_t)bh * 65536;   // [32][2048] transposed
  const half_t* Vh  = Pv + (size_t)bh * 65536;   // [2048][32] natural
  const float* q0p = q0a + bh * SS;
  const float* k0p = k0a + bh * SS;
  const float* tvp = tva + bh * SS;

  int qrow0 = blockIdx.x * 64 + wave * 16;

  // A (query) tile, resident for the whole key loop
  const half_t* ar = Qh + (size_t)(qrow0 + mm) * 32;
  v16h aq = cat8(*(const v8h*)(ar + 8 * hg), *(const v8h*)(ar + 16 + 8 * hg));

  float q0r[8];
#pragma unroll
  for (int r = 0; r < 8; ++r) q0r[r] = q0p[qrow0 + 8 * hg + r];

  float bet = betap[0], cc = cparp[0];
  float den[8];
#pragma unroll
  for (int r = 0; r < 8; ++r) den[r] = 0.0f;
  v8f cn0 = {}, cn1 = {};
  const v8f zc = {};
  half_t* at = &atile[wave][0];

#if HAVE_TDM
  if (wave == 0) {  // prologue: DMA tile 0 into buffer 0
    tdm_load_2d_f16((unsigned)(uintptr_t)&ktile[0][0], Kth,
                    SS, 32, 64, 32, SS);
    tdm_load_2d_f16((unsigned)(uintptr_t)&vtile[0][0], Vh,
                    64 * 32, 1, 64 * 32, 1, 64 * 32);
  }
#endif

  for (int it = 0; it < SS / 64; ++it) {
    int kt = it * 64;
#if HAVE_TDM
    if (wave == 0) {
      if (it + 1 < SS / 64) {   // prefetch next tile into the other buffer
        int ktn = kt + 64;
        tdm_load_2d_f16((unsigned)(uintptr_t)&ktile[(it + 1) & 1][0],
                        Kth + ktn, SS, 32, 64, 32, SS);
        tdm_load_2d_f16((unsigned)(uintptr_t)&vtile[(it + 1) & 1][0],
                        Vh + (size_t)ktn * 32, 64 * 32, 1, 64 * 32, 1, 64 * 32);
        __builtin_amdgcn_s_wait_tensorcnt(2);   // current tile's 2 DMAs done
      } else {
        __builtin_amdgcn_s_wait_tensorcnt(0);
      }
    }
    __syncthreads();            // current buffer visible to all waves
    const half_t* kbuf = &ktile[it & 1][0];
    const half_t* vbuf = &vtile[it & 1][0];
#else
    __syncthreads();
    {   // cooperative fallback copy, single buffer
      int e = tid * 16;                        // 2048 elems, 16 per thread
      int d = e >> 6, c = e & 63;
      const half_t* ks = Kth + (size_t)d * SS + kt + c;
      half_t* kd = &ktile[0][e];
#pragma unroll
      for (int i = 0; i < 16; ++i) kd[i] = ks[i];
      const half_t* vs = Vh + (size_t)kt * 32 + e;
      half_t* vd = &vtile[0][e];
#pragma unroll
      for (int i = 0; i < 16; ++i) vd[i] = vs[i];
    }
    __syncthreads();
    const half_t* kbuf = &ktile[0][0];
    const half_t* vbuf = &vtile[0][0];
#endif

    // ---- scores: 4 x (16q x 16k) WMMAs over K=32 Lorentz dims ----
    v8f c[4];
#pragma unroll
    for (int t = 0; t < 4; ++t) {
      v16h bk;  // lane<->key, elems<->d
#pragma unroll
      for (int i = 0; i < 8; ++i) {
        bk[i]     = kbuf[(8 * hg + i) * 64 + t * 16 + mm];
        bk[i + 8] = kbuf[(16 + 8 * hg + i) * 64 + t * 16 + mm];
      }
      c[t] = __builtin_amdgcn_wmma_f32_16x16x32_f16(false, aq, false, bk,
                                                    (short)0, zc, false, false);
    }
    float k0v[4], tvv[4];
#pragma unroll
    for (int t = 0; t < 4; ++t) {
      k0v[t] = k0p[kt + t * 16 + mm];
      tvv[t] = tvp[kt + t * 16 + mm];
    }
    // ---- alpha = exp(-beta*arccosh(max(q0k0 - q.k, 1+eps)) - c) ----
#pragma unroll
    for (int t = 0; t < 4; ++t) {
#pragma unroll
      for (int r = 0; r < 8; ++r) {
        float s0 = fmaf(q0r[r], k0v[t], -c[t][r]);
        float x0 = fmaxf(s0, 1.0f + 1e-6f);
        float d0 = __logf(x0 + __builtin_sqrtf(fmaf(x0, x0, -1.0f)));
        float a0 = __expf(fmaf(-bet, d0, -cc));
        den[r] = fmaf(a0, tvv[t], den[r]);
        at[(r + 8 * hg) * 64 + t * 16 + mm] = (half_t)a0;
      }
    }
    asm volatile("s_wait_dscnt 0" ::: "memory");  // same-wave LDS RAW

    // ---- midpoint numerator: alpha[16x64] @ V[64x32] as 2 halves ----
#pragma unroll
    for (int hf = 0; hf < 2; ++hf) {
      const half_t* ar2 = at + mm * 64 + hf * 32;
      v16h aal = cat8(*(const v8h*)(ar2 + 8 * hg), *(const v8h*)(ar2 + 16 + 8 * hg));
      const half_t* vr = vbuf + (size_t)(hf * 32) * 32 + mm;  // lane<->vd, elems<->key
      v16h bv0, bv1;
#pragma unroll
      for (int i = 0; i < 8; ++i) {
        bv0[i]     = vr[(8 * hg + i) * 32];
        bv0[i + 8] = vr[(16 + 8 * hg + i) * 32];
        bv1[i]     = vr[(8 * hg + i) * 32 + 16];
        bv1[i + 8] = vr[(16 + 8 * hg + i) * 32 + 16];
      }
      cn0 = __builtin_amdgcn_wmma_f32_16x16x32_f16(false, aal, false,
              bv0, (short)0, cn0, false, false);
      cn1 = __builtin_amdgcn_wmma_f32_16x16x32_f16(false, aal, false,
              bv1, (short)0, cn1, false, false);
    }
#if HAVE_TDM
    __syncthreads();            // all waves done reading before DMA overwrites
#endif
  }

  // Reduce den over the 16 lanes that share each M row (xor toggles mm bits only)
#pragma unroll
  for (int r = 0; r < 8; ++r) {
    float d = den[r];
    d += __shfl_xor(d, 1, 32);
    d += __shfl_xor(d, 2, 32);
    d += __shfl_xor(d, 4, 32);
    d += __shfl_xor(d, 8, 32);
    den[r] = fmaxf(d, 1e-6f);
  }

  // res[b][s][h*32 + vd] = num/den  (swapaxes+reshape of the reference)
  size_t rbase = (size_t)b * SS * HD;
#pragma unroll
  for (int r = 0; r < 8; ++r) {
    int s = qrow0 + r + 8 * hg;
    float inv = 1.0f / den[r];
    res[rbase + (size_t)s * HD + h * 32 + mm]      = (half_t)(cn0[r] * inv);
    res[rbase + (size_t)s * HD + h * 32 + 16 + mm] = (half_t)(cn1[r] * inv);
  }
}

// ---------------------------------------------------------------------------
// Output projection: out = res @ Wo^T + bo  ([4096,256] x [256,512] -> f32)
// grid = (64, 8), block = 128.
// ---------------------------------------------------------------------------
__global__ void __launch_bounds__(128)
gemm_out_kernel(const half_t* __restrict__ Rm, const half_t* __restrict__ WoT,
                const float* __restrict__ bo, float* __restrict__ out) {
  int tid = threadIdx.x, wave = tid >> 5, lane = tid & 31;
  int mm = lane & 15, hg = lane >> 4;
  int m0 = blockIdx.x * 64 + wave * 16;
  int n0 = blockIdx.y * 64;

  v8f acc[4] = {};
  for (int k0 = 0; k0 < HD; k0 += 32) {
    const half_t* ar = Rm + (size_t)(m0 + mm) * HD + k0;
    v16h a = cat8(*(const v8h*)(ar + 8 * hg), *(const v8h*)(ar + 16 + 8 * hg));
    const half_t* br = WoT + (size_t)k0 * DM + n0 + mm;
#pragma unroll
    for (int t = 0; t < 4; ++t) {
      v16h b;
#pragma unroll
      for (int i = 0; i < 8; ++i) {
        b[i]     = br[(size_t)(8 * hg + i) * DM + t * 16];
        b[i + 8] = br[(size_t)(16 + 8 * hg + i) * DM + t * 16];
      }
      acc[t] = __builtin_amdgcn_wmma_f32_16x16x32_f16(false, a, false, b,
                                                      (short)0, acc[t], false, false);
    }
  }
#pragma unroll
  for (int t = 0; t < 4; ++t) {
    int col = n0 + t * 16 + mm;
    float bb = bo[col];
#pragma unroll
    for (int r = 0; r < 8; ++r) {
      int row = m0 + r + 8 * hg;
      out[(size_t)row * DM + col] = acc[t][r] + bb;
    }
  }
}

// ---------------------------------------------------------------------------
extern "C" void kernel_launch(void* const* d_in, const int* in_sizes, int n_in,
                              void* d_out, int out_size, void* d_ws, size_t ws_size,
                              hipStream_t stream) {
  const float* q    = (const float*)d_in[0];
  const float* k    = (const float*)d_in[1];
  const float* v    = (const float*)d_in[2];
  const float* Wq_w = (const float*)d_in[3];
  const float* Wq_b = (const float*)d_in[4];
  const float* Wk_w = (const float*)d_in[5];
  const float* Wk_b = (const float*)d_in[6];
  const float* Wv_w = (const float*)d_in[7];
  const float* Wv_b = (const float*)d_in[8];
  const float* Wo_w = (const float*)d_in[9];
  const float* Wo_b = (const float*)d_in[10];
  const float* beta = (const float*)d_in[11];
  const float* cpar = (const float*)d_in[12];
  float* out = (float*)d_out;

  // Workspace carve-up (~11.7 MB total)
  char* w = (char*)d_ws;
  size_t off = 0;
  auto carve = [&](size_t bytes) { char* p = w + off; off += (bytes + 255) & ~(size_t)255; return p; };
  half_t* Pq  = (half_t*)carve((size_t)NROW * HD * 2);
  half_t* Pk  = (half_t*)carve((size_t)NROW * HD * 2);
  half_t* Pv  = (half_t*)carve((size_t)NROW * HD * 2);
  half_t* Kt  = (half_t*)carve((size_t)BB * HH * 32 * SS * 2);
  half_t* WqT = (half_t*)carve((size_t)DM * HD * 2);
  half_t* WkT = (half_t*)carve((size_t)DM * HD * 2);
  half_t* WvT = (half_t*)carve((size_t)DM * HD * 2);
  half_t* WoT = (half_t*)carve((size_t)HD * DM * 2);
  float*  q0  = (float*)carve((size_t)HEADROWS * 4);
  float*  k0  = (float*)carve((size_t)HEADROWS * 4);
  float*  tv  = (float*)carve((size_t)HEADROWS * 4);
  half_t* res = (half_t*)carve((size_t)NROW * HD * 2);

  // 1) Weight transposes (f32 -> f16, [out,in] -> [in,out])
  int wn = HD * DM;
  wtrans_kernel<<<dim3((wn + 255) / 256), 256, 0, stream>>>(Wq_w, WqT, HD, DM);
  wtrans_kernel<<<dim3((wn + 255) / 256), 256, 0, stream>>>(Wk_w, WkT, HD, DM);
  wtrans_kernel<<<dim3((wn + 255) / 256), 256, 0, stream>>>(Wv_w, WvT, HD, DM);
  wtrans_kernel<<<dim3((wn + 255) / 256), 256, 0, stream>>>(Wo_w, WoT, DM, HD);

  // 2) Q/K/V projections (WMMA)
  gemm_proj_kernel<<<dim3(NROW / 64, HD / 64, 3), 128, 0, stream>>>(
      q, k, v, WqT, WkT, WvT, Wq_b, Wk_b, Wv_b, Pq, Pk, Pv);

  // 3) q0/k0/t_v norms + per-head K transpose
  aux_norm_kernel<<<dim3(3 * HEADROWS / 256), 256, 0, stream>>>(Pq, Pk, Pv, q0, k0, tv);
  ktrans_kernel<<<dim3((BB * HH * 32 * SS) / 256), 256, 0, stream>>>(Pk, Kt);

  // 4) Fused attention (TDM-fed, WMMA scores + WMMA midpoint)
  attn_kernel<<<dim3(SS / 64, BB * HH), 128, 0, stream>>>(
      Pq, Kt, Pv, q0, k0, tv, beta, cpar, res);

  // 5) Output projection (WMMA, f32 out)
  gemm_out_kernel<<<dim3(NROW / 64, DM / 64), 128, 0, stream>>>(res, WoT, Wo_b, out);
}
